// NonLocalBlock3D_12627203850615
// MI455X (gfx1250) — matrix operand order repaired and checked by reading, not compiled, not run on hardware
//
#include <hip/hip_runtime.h>
#include <stdint.h>

#define NN    8192
#define CDIM  256
#define IDIM  128
#define SPLIT 4
#define KEYS_PER_SPLIT (NN / SPLIT)   // 2048

typedef __attribute__((ext_vector_type(16))) __bf16 v16bf;
typedef __attribute__((ext_vector_type(8)))  float  v8f;

// ---------------------------------------------------------------- helpers

static __device__ __forceinline__ __bf16 f2bf(float f) {
    return (__bf16)f;                 // backend: v_cvt_pk_bf16_f32 path
}

// A-fragment (16x32 bf16, M x K): lane L -> row row0+(L&15);
// g = L>>4 selects K chunks [g*8, g*8+8) and [g*8+16, g*8+24).
static __device__ __forceinline__ v16bf load_a_frag(const __bf16* base, int stride,
                                                    int row0, int lane, int kb) {
    const int r = row0 + (lane & 15);
    const int g = lane >> 4;
    const __bf16* p = base + (long)r * stride + kb + g * 8;
    v16bf a;
#pragma unroll
    for (int i = 0; i < 8; i++) a[i] = p[i];
#pragma unroll
    for (int i = 0; i < 8; i++) a[8 + i] = p[16 + i];
    return a;
}

// B-fragment (32x16 bf16, K x N) with B stored "N-major" (row n holds K-contiguous
// data): lane L -> column n0+(L&15); g = L>>4 selects K chunks [g*8, g*8+8) and [g*8+16, g*8+24).
static __device__ __forceinline__ v16bf load_b_frag(const __bf16* base, int stride,
                                                    int n0, int lane, int kb) {
    const int n = n0 + (lane & 15);
    const int g = lane >> 4;
    const __bf16* p = base + (long)n * stride + kb + g * 8;
    v16bf b;
#pragma unroll
    for (int i = 0; i < 16; i++) b[i] = p[(i & 7) + ((i >> 3) << 4)];
    return b;
}

static __device__ __forceinline__ v8f wmma_bf16(v16bf a, v16bf b, v8f c) {
    return __builtin_amdgcn_wmma_f32_16x16x32_bf16(false, a, false, b,
                                                   (short)0, c, false, false);
}

// low 32 bits of a generic pointer to LDS == wave-relative LDS byte address
static __device__ __forceinline__ uint32_t lds_addr(const void* p) {
    return (uint32_t)(uintptr_t)p;
}

static __device__ __forceinline__ void async_ld_b128(uint32_t lds, uint64_t gaddr) {
    asm volatile("global_load_async_to_lds_b128 %0, %1, off"
                 :: "v"(lds), "v"(gaddr) : "memory");
}

// ---------------------------------------------------------------- converts

__global__ __launch_bounds__(256) void cvt_kernel(const float* __restrict__ src,
                                                  uint16_t* __restrict__ dst_, int n) {
    __bf16* dst = (__bf16*)dst_;
    int i = blockIdx.x * blockDim.x + threadIdx.x;
    if (i < n) dst[i] = f2bf(src[i]);
}

// src [R][S] f32  ->  dst [S][R] bf16 (transpose + convert)
__global__ __launch_bounds__(256) void cvtT_kernel(const float* __restrict__ src,
                                                   uint16_t* __restrict__ dst_,
                                                   int R, int S) {
    __bf16* dst = (__bf16*)dst_;
    int i = blockIdx.x * blockDim.x + threadIdx.x;
    if (i < R * S) {
        int r = i / S, s = i % S;
        dst[(long)s * R + r] = f2bf(src[i]);
    }
}

// ---------------------------------------------------------------- projections
// theta/phi: row-major [NN][IDIM] bf16.  g: transposed [IDIM][NN] bf16.

__global__ __launch_bounds__(128) void proj_kernel(
        const uint16_t* __restrict__ xh_,
        const uint16_t* __restrict__ wt_th_, const float* __restrict__ b_th,
        const uint16_t* __restrict__ wt_ph_, const float* __restrict__ b_ph,
        const uint16_t* __restrict__ wt_g_,  const float* __restrict__ b_g,
        uint16_t* __restrict__ theta_, uint16_t* __restrict__ phi_,
        uint16_t* __restrict__ gT_) {
    const __bf16* xh = (const __bf16*)xh_;
    const int lane = threadIdx.x & 31;
    const int wid  = blockIdx.x * (blockDim.x >> 5) + (threadIdx.x >> 5);
    const int proj = wid / 4096;
    const int rem  = wid % 4096;
    const int row0 = (rem >> 3) * 16;
    const int col0 = (rem & 7) * 16;

    const __bf16* wt  = (proj == 0) ? (const __bf16*)wt_th_
                       : (proj == 1) ? (const __bf16*)wt_ph_
                                     : (const __bf16*)wt_g_;
    const float* bias = (proj == 0) ? b_th : (proj == 1) ? b_ph : b_g;

    v8f acc = {};
#pragma unroll
    for (int kb = 0; kb < CDIM; kb += 32) {
        v16bf a = load_a_frag(xh, CDIM, row0, lane, kb);
        v16bf b = load_b_frag(wt, CDIM, col0, lane, kb);
        acc = wmma_bf16(a, b, acc);
    }

    const float bv = bias[col0 + (lane & 15)];
    const int h = lane >> 4, col = col0 + (lane & 15);
    if (proj < 2) {
        __bf16* dst = (proj == 0) ? (__bf16*)theta_ : (__bf16*)phi_;
#pragma unroll
        for (int r = 0; r < 8; r++)
            dst[(long)(row0 + r + h * 8) * IDIM + col] = f2bf(acc[r] + bv);
    } else {
        __bf16* gT = (__bf16*)gT_;
#pragma unroll
        for (int r = 0; r < 8; r++)
            gT[(long)col * NN + row0 + r + h * 8] = f2bf(acc[r] + bv);
    }
}

// ---------------------------------------------------------------- fused attention
// Key-split flash attention. Block = 4 waves = 64 query rows; each block handles
// KEYS_PER_SPLIT keys. K/V tiles double-buffered into LDS via async-to-LDS DMA.
// Emits unnormalized partial O (f32) plus per-row (m, l) for the combine pass.

__global__ __launch_bounds__(128) void attn_kernel(
        const uint16_t* __restrict__ theta_, const uint16_t* __restrict__ phi_,
        const uint16_t* __restrict__ gT_,
        float* __restrict__ Opart, float* __restrict__ ml) {
    const __bf16* theta = (const __bf16*)theta_;
    const __bf16* phi   = (const __bf16*)phi_;
    const __bf16* gT    = (const __bf16*)gT_;

    __shared__ __bf16 kbuf[2][32 * IDIM];   // 32 keys x 128 feats (key-major)
    __shared__ __bf16 vbuf[2][IDIM * 32];   // 128 feats x 32 keys (feat-major)
    __shared__ __bf16 pbuf[4][16 * 32];     // per-wave P staging

    const int lane  = threadIdx.x & 31;
    const int wave  = threadIdx.x >> 5;
    const int t     = threadIdx.x;
    const int rb    = blockIdx.x & 127;           // 128 row-blocks of 64 rows
    const int split = blockIdx.x >> 7;            // SPLIT key slices
    const int row0  = rb * 64 + wave * 16;
    const int kbase = split * KEYS_PER_SPLIT;

    const uint32_t klds[2] = { lds_addr(&kbuf[0][0]), lds_addr(&kbuf[1][0]) };
    const uint32_t vlds[2] = { lds_addr(&vbuf[0][0]), lds_addr(&vbuf[1][0]) };

    // stage 32 keys of phi (contiguous 8KB) + gT columns (128 rows x 64B) into LDS.
    // 8 async b128 issues per thread -> wave ASYNCcnt += 8 per stage.
    auto stage = [&](int buf, int kb) {
        const uint64_t gp = (uint64_t)(uintptr_t)phi + (uint64_t)kb * IDIM * 2;
#pragma unroll
        for (int i = 0; i < 4; i++)
            async_ld_b128(klds[buf] + (uint32_t)(i * 128 + t) * 16,
                          gp + (uint64_t)(i * 128 + t) * 16);
        const uint64_t gv = (uint64_t)(uintptr_t)gT + ((uint64_t)t * NN + kb) * 2;
        const uint32_t lv = vlds[buf] + (uint32_t)t * 64;
#pragma unroll
        for (int i = 0; i < 4; i++)
            async_ld_b128(lv + (uint32_t)i * 16, gv + (uint64_t)i * 16);
    };

    // Q fragments: K = 128 features -> 4 fragments of K=32
    v16bf qf[4];
#pragma unroll
    for (int i = 0; i < 4; i++) qf[i] = load_a_frag(theta, IDIM, row0, lane, i * 32);

    v8f acc[8];
#pragma unroll
    for (int ft = 0; ft < 8; ft++) acc[ft] = v8f{};
    float m[8], l[8];
#pragma unroll
    for (int r = 0; r < 8; r++) { m[r] = -1e30f; l[r] = 0.0f; }

    __bf16* pl = pbuf[wave];
    const int h = lane >> 4, col = lane & 15;
    const int nIt = KEYS_PER_SPLIT / 32;

    stage(0, kbase);
    for (int it = 0; it < nIt; ++it) {
        const int buf = it & 1;
        if (it + 1 < nIt) {
            stage(buf ^ 1, kbase + (it + 1) * 32);
            asm volatile("s_wait_asynccnt 0x8" ::: "memory");   // buf's group done
        } else {
            asm volatile("s_wait_asynccnt 0x0" ::: "memory");
        }
        __syncthreads();

        const __bf16* kb_p = &kbuf[buf][0];
        const __bf16* vb_p = &vbuf[buf][0];

        // ---- scores S = theta * phi^T for 16 rows x 32 keys (two 16-key tiles)
        v8f s0 = {}, s1 = {};
#pragma unroll
        for (int i = 0; i < 4; i++) {
            v16bf b0 = load_b_frag(kb_p, IDIM, 0,  lane, i * 32);
            s0 = wmma_bf16(qf[i], b0, s0);
        }
#pragma unroll
        for (int i = 0; i < 4; i++) {
            v16bf b1 = load_b_frag(kb_p, IDIM, 16, lane, i * 32);
            s1 = wmma_bf16(qf[i], b1, s1);
        }

        // ---- online softmax: rows live across the 16 lanes of each half-wave
#pragma unroll
        for (int r = 0; r < 8; r++) {
            float bm = fmaxf(s0[r], s1[r]);
            bm = fmaxf(bm, __shfl_xor(bm, 1));
            bm = fmaxf(bm, __shfl_xor(bm, 2));
            bm = fmaxf(bm, __shfl_xor(bm, 4));
            bm = fmaxf(bm, __shfl_xor(bm, 8));
            const float mn = fmaxf(m[r], bm);
            const float sc = __expf(m[r] - mn);
            const float p0 = __expf(s0[r] - mn);
            const float p1 = __expf(s1[r] - mn);
            float rs = p0 + p1;
            rs += __shfl_xor(rs, 1);
            rs += __shfl_xor(rs, 2);
            rs += __shfl_xor(rs, 4);
            rs += __shfl_xor(rs, 8);
            l[r] = l[r] * sc + rs;
            m[r] = mn;
            s0[r] = p0;
            s1[r] = p1;
#pragma unroll
            for (int ft = 0; ft < 8; ft++) acc[ft][r] *= sc;
        }

        // ---- D-layout -> A-layout for P via per-wave LDS (same-wave DS in order)
#pragma unroll
        for (int r = 0; r < 8; r++) {
            pl[(r + h * 8) * 32 + col]      = f2bf(s0[r]);
            pl[(r + h * 8) * 32 + 16 + col] = f2bf(s1[r]);
        }
        v16bf pa = load_a_frag(pl, 32, 0, lane, 0);

        // ---- O += P * V  (vbuf is feat-major, 32 local keys per row)
#pragma unroll
        for (int ft = 0; ft < 8; ft++) {
            v16bf vb = load_b_frag(vb_p, 32, ft * 16, lane, 0);
            acc[ft] = wmma_bf16(pa, vb, acc[ft]);
        }
        __syncthreads();    // all waves done reading buf before it is restaged
    }

    // ---- emit unnormalized partial O + per-row stats
#pragma unroll
    for (int r = 0; r < 8; r++) {
        const int row = row0 + r + h * 8;
        const long obase = ((long)split * NN + row) * IDIM;
#pragma unroll
        for (int ft = 0; ft < 8; ft++)
            Opart[obase + ft * 16 + col] = acc[ft][r];
        if (col == 0) {
            ml[((long)split * NN + row) * 2]     = m[r];
            ml[((long)split * NN + row) * 2 + 1] = l[r];
        }
    }
}

// ---------------------------------------------------------------- lse combine
// y[row] = sum_s exp(m_s - m*) O_s[row] / sum_s l_s exp(m_s - m*)

__global__ __launch_bounds__(256) void comb_kernel(
        const float* __restrict__ Opart, const float* __restrict__ ml,
        uint16_t* __restrict__ yh_) {
    __bf16* yh = (__bf16*)yh_;
    const int i = blockIdx.x * blockDim.x + threadIdx.x;   // over NN*IDIM
    const int row = i >> 7;
    float mx = -1e30f;
#pragma unroll
    for (int s = 0; s < SPLIT; s++)
        mx = fmaxf(mx, ml[((long)s * NN + row) * 2]);
    float L = 0.0f, acc = 0.0f;
#pragma unroll
    for (int s = 0; s < SPLIT; s++) {
        const float ms = ml[((long)s * NN + row) * 2];
        const float ls = ml[((long)s * NN + row) * 2 + 1];
        const float w  = __expf(ms - mx);
        L   += ls * w;
        acc += w * Opart[(long)s * NN * IDIM + i];
    }
    yh[i] = f2bf(acc / L);
}

// ---------------------------------------------------------------- output proj
// out = y @ W_w + W_b + x   (f32 output)

__global__ __launch_bounds__(128) void out_kernel(
        const uint16_t* __restrict__ yh_, const uint16_t* __restrict__ wwT_,
        const float* __restrict__ wb, const float* __restrict__ x,
        float* __restrict__ out) {
    const __bf16* yh  = (const __bf16*)yh_;
    const __bf16* wwT = (const __bf16*)wwT_;
    const int lane = threadIdx.x & 31;
    const int wid  = blockIdx.x * (blockDim.x >> 5) + (threadIdx.x >> 5);
    const int row0 = (wid >> 4) * 16;
    const int col0 = (wid & 15) * 16;

    v8f acc = {};
#pragma unroll
    for (int kb = 0; kb < IDIM; kb += 32) {
        v16bf a = load_a_frag(yh, IDIM, row0, lane, kb);
        v16bf b = load_b_frag(wwT, IDIM, col0, lane, kb);
        acc = wmma_bf16(a, b, acc);
    }
    const float bv = wb[col0 + (lane & 15)];
    const int h = lane >> 4;
#pragma unroll
    for (int r = 0; r < 8; r++) {
        const long idx = (long)(row0 + r + h * 8) * CDIM + col0 + (lane & 15);
        out[idx] = acc[r] + bv + x[idx];
    }
}

// ---------------------------------------------------------------- launch

extern "C" void kernel_launch(void* const* d_in, const int* in_sizes, int n_in,
                              void* d_out, int out_size, void* d_ws, size_t ws_size,
                              hipStream_t stream) {
    const float* x       = (const float*)d_in[0];
    const float* g_w     = (const float*)d_in[1];
    const float* g_b     = (const float*)d_in[2];
    const float* theta_w = (const float*)d_in[3];
    const float* theta_b = (const float*)d_in[4];
    const float* phi_w   = (const float*)d_in[5];
    const float* phi_b   = (const float*)d_in[6];
    const float* W_w     = (const float*)d_in[7];
    const float* W_b     = (const float*)d_in[8];
    float* out = (float*)d_out;

    char* base = (char*)d_ws;
    size_t off = 0;
    auto take = [&](size_t bytes) { char* p = base + off;
                                    off = (off + bytes + 255) & ~(size_t)255;
                                    return (void*)p; };
    uint16_t* xh    = (uint16_t*)take((size_t)NN * CDIM * 2);
    uint16_t* thh   = (uint16_t*)take((size_t)NN * IDIM * 2);
    uint16_t* phh   = (uint16_t*)take((size_t)NN * IDIM * 2);
    uint16_t* gT    = (uint16_t*)take((size_t)IDIM * NN * 2);
    uint16_t* yhh   = (uint16_t*)take((size_t)NN * IDIM * 2);
    uint16_t* wt_g  = (uint16_t*)take((size_t)IDIM * CDIM * 2);
    uint16_t* wt_th = (uint16_t*)take((size_t)IDIM * CDIM * 2);
    uint16_t* wt_ph = (uint16_t*)take((size_t)IDIM * CDIM * 2);
    uint16_t* wwT   = (uint16_t*)take((size_t)CDIM * IDIM * 2);
    float*    Opart = (float*)take((size_t)SPLIT * NN * IDIM * 4);  // 16 MB
    float*    mlbuf = (float*)take((size_t)SPLIT * NN * 2 * 4);
    (void)in_sizes; (void)n_in; (void)out_size; (void)ws_size;

    // 1) converts
    cvt_kernel<<<(NN * CDIM + 255) / 256, 256, 0, stream>>>(x, xh, NN * CDIM);
    cvtT_kernel<<<(CDIM * IDIM + 255) / 256, 256, 0, stream>>>(g_w,     wt_g,  CDIM, IDIM);
    cvtT_kernel<<<(CDIM * IDIM + 255) / 256, 256, 0, stream>>>(theta_w, wt_th, CDIM, IDIM);
    cvtT_kernel<<<(CDIM * IDIM + 255) / 256, 256, 0, stream>>>(phi_w,   wt_ph, CDIM, IDIM);
    cvtT_kernel<<<(IDIM * CDIM + 255) / 256, 256, 0, stream>>>(W_w,     wwT,   IDIM, CDIM);

    // 2) projections: 3 * 4096 tiles, 1 tile/wave, 4 waves/block
    proj_kernel<<<3 * 4096 / 4, 128, 0, stream>>>(xh, wt_th, theta_b, wt_ph, phi_b,
                                                  wt_g, g_b, thh, phh, gT);

    // 3) key-split flash attention: SPLIT * 128 blocks, 4 waves each
    attn_kernel<<<SPLIT * 128, 128, 0, stream>>>(thh, phh, gT, Opart, mlbuf);

    // 4) log-sum-exp combine
    comb_kernel<<<(NN * IDIM + 255) / 256, 256, 0, stream>>>(Opart, mlbuf, yhh);

    // 5) output projection + bias + residual
    out_kernel<<<8192 / 4, 128, 0, stream>>>(yhh, wwT, W_b, x, out);
}
